// ST_ViT_85581518340412
// MI455X (gfx1250) — compile-verified, matrix operand order checked by reading, not run on hardware
//
#include <hip/hip_runtime.h>

typedef __attribute__((ext_vector_type(16))) _Float16 v16h;
typedef __attribute__((ext_vector_type(4)))  _Float16 v4h;
typedef __attribute__((ext_vector_type(8)))  float    v8f;

#define DIMC   128
#define NTOK   3072      // T*P
#define ROWS   6144      // B*NTOK
#define NHEAD  8
#define DHEAD  16
#define MLPD   512
#define NDEPTH 8
#define NKT    192       // n/16 key tiles
#define LPAD   8         // LDS row pad (halves) -> conflict-free B-frag reads
#define ATT_SCALE 0.08838834764831845f   // 128^-0.5

// ---------------- reductions (wave32; xor masks < 16 stay inside a half-wave)
__device__ __forceinline__ float rmax16(float v) {
  v = fmaxf(v, __shfl_xor(v, 1));
  v = fmaxf(v, __shfl_xor(v, 2));
  v = fmaxf(v, __shfl_xor(v, 4));
  v = fmaxf(v, __shfl_xor(v, 8));
  return v;
}
__device__ __forceinline__ float rsum16(float v) {
  v += __shfl_xor(v, 1);
  v += __shfl_xor(v, 2);
  v += __shfl_xor(v, 4);
  v += __shfl_xor(v, 8);
  return v;
}
__device__ __forceinline__ float rsum32(float v) {
  v += __shfl_xor(v, 1);
  v += __shfl_xor(v, 2);
  v += __shfl_xor(v, 4);
  v += __shfl_xor(v, 8);
  v += __shfl_xor(v, 16);
  return v;
}

// ---------------- async global->LDS copy (ASYNCcnt-tracked, CDNA5)
__device__ __forceinline__ void async_g2l_b128(const _Float16* g, _Float16* l) {
  unsigned loff = (unsigned)(size_t)l;                 // low 32 bits = LDS offset
  unsigned long long ga = (unsigned long long)(size_t)g;
  asm volatile("global_load_async_to_lds_b128 %0, %1, off"
               :: "v"(loff), "v"(ga) : "memory");
}
__device__ __forceinline__ void wait_async0() {
  asm volatile("s_wait_asynccnt 0x0" ::: "memory");
}

// ---------------- GEMM strip mainloop: one wave -> 16x64 output (4 WMMA chains).
// A: [M,K] f16 row-major. Wt: [N,K] f16 (weights pre-transposed).
// STAGE: block cooperatively stages the 64xK weight strip into padded LDS
// via async loads; all 4 waves reuse it (B-frags then come from LDS banks).
// B-fragments are loaded into 4 distinct variables before the 4 WMMAs so the
// scheduler can clause the ds_loads and use partial s_wait_dscnt thresholds.
template<int K, bool STAGE>
__device__ __forceinline__ void gemm_strip(const _Float16* __restrict__ A,
                                           const _Float16* __restrict__ Wt,
                                           int m0, int n0s, _Float16* bsh,
                                           v8f acc[4]) {
  const int tid  = threadIdx.x;
  const int lane = tid & 31;
  const int rr   = lane & 15;
  const int kb   = (lane >> 4) << 4;

  if (STAGE) {
    const _Float16* gsrc = Wt + (size_t)n0s * K;
    const int cpr = K / 8;                       // 16B chunks per row
    for (int i = tid; i < 64 * cpr; i += 128) {  // 128 threads per block
      int rw = i / cpr, cx = i - rw * cpr;
      async_g2l_b128(gsrc + (size_t)rw * K + cx * 8, bsh + rw * (K + LPAD) + cx * 8);
    }
    wait_async0();
    __syncthreads();
  }

  const _Float16* ap = A + (size_t)(m0 + rr) * K + kb;
  const int ldb = STAGE ? (K + LPAD) : K;
  const _Float16* bbase = STAGE ? (const _Float16*)bsh : (Wt + (size_t)n0s * K);
  bbase += (size_t)rr * ldb + kb;

  acc[0] = (v8f){}; acc[1] = (v8f){}; acc[2] = (v8f){}; acc[3] = (v8f){};
  for (int k = 0; k < K; k += 32) {
    v16h a  = *(const v16h*)(ap + k);
    v16h b0 = *(const v16h*)(bbase + (size_t)(0 * 16) * ldb + k);
    v16h b1 = *(const v16h*)(bbase + (size_t)(1 * 16) * ldb + k);
    v16h b2 = *(const v16h*)(bbase + (size_t)(2 * 16) * ldb + k);
    v16h b3 = *(const v16h*)(bbase + (size_t)(3 * 16) * ldb + k);
    acc[0] = __builtin_amdgcn_wmma_f32_16x16x32_f16(false, a, false, b0, (short)0, acc[0], false, false);
    acc[1] = __builtin_amdgcn_wmma_f32_16x16x32_f16(false, a, false, b1, (short)0, acc[1], false, false);
    acc[2] = __builtin_amdgcn_wmma_f32_16x16x32_f16(false, a, false, b2, (short)0, acc[2], false, false);
    acc[3] = __builtin_amdgcn_wmma_f32_16x16x32_f16(false, a, false, b3, (short)0, acc[3], false, false);
  }
}

// ---------------- weight convert + transpose: fp32 [L][K][N] -> f16 [L][N][K]
__global__ void k_wconv_t(const float* __restrict__ src, _Float16* __restrict__ dst,
                          int K, int N, int total) {
  int idx = blockIdx.x * 256 + threadIdx.x;
  if (idx >= total) return;
  int l   = idx / (K * N);
  int rem = idx - l * K * N;
  int k   = rem / N;
  int c   = rem - k * N;
  dst[(size_t)l * K * N + (size_t)c * K + k] = (_Float16)src[idx];
}

// ---------------- patch embed
__global__ void k_patch_embed(const float* __restrict__ x, const float* __restrict__ pos,
                              const float* __restrict__ tim, const float* __restrict__ wp,
                              const float* __restrict__ bp, float* __restrict__ h) {
  int row = blockIdx.x;       // 0..6143
  int d   = threadIdx.x;      // 0..127
  int b   = row / NTOK;
  int tp  = row - b * NTOK;
  int t   = tp >> 8;
  int p   = tp & 255;
  float acc = bp[d] + pos[p * DIMC + d] + tim[t * DIMC + d];
#pragma unroll
  for (int c = 0; c < 7; ++c)
    acc += x[(((b * 12 + t) * 7 + c) << 8) + p] * wp[c * DIMC + d];
  h[(size_t)row * DIMC + d] = acc;
}

// ---------------- layernorm: one wave per row, float4 loads, shfl reduce
__global__ void k_ln(const float* __restrict__ h, const float* __restrict__ g,
                     const float* __restrict__ bb, _Float16* __restrict__ out) {
  int wave = threadIdx.x >> 5, lane = threadIdx.x & 31;
  int row  = blockIdx.x * 8 + wave;
  float4 v = ((const float4*)(h + (size_t)row * DIMC))[lane];
  float s = v.x + v.y + v.z + v.w;
  float q = v.x * v.x + v.y * v.y + v.z * v.z + v.w * v.w;
  s = rsum32(s); q = rsum32(q);
  float mu  = s * (1.0f / DIMC);
  float var = q * (1.0f / DIMC) - mu * mu;
  float rs  = rsqrtf(var + 1e-5f);
  float4 gg = ((const float4*)g)[lane];
  float4 bv = ((const float4*)bb)[lane];
  v4h o;
  o[0] = (_Float16)((v.x - mu) * rs * gg.x + bv.x);
  o[1] = (_Float16)((v.y - mu) * rs * gg.y + bv.y);
  o[2] = (_Float16)((v.z - mu) * rs * gg.z + bv.z);
  o[3] = (_Float16)((v.w - mu) * rs * gg.w + bv.w);
  *(v4h*)(out + (size_t)row * DIMC + lane * 4) = o;
}

// ---------------- QKV GEMM: strip-uniform Q/K/V select, scalar batch index
__global__ void k_gemm_qkv(const _Float16* __restrict__ A, const _Float16* __restrict__ Wt,
                           _Float16* __restrict__ Qh, _Float16* __restrict__ Kh,
                           _Float16* __restrict__ Vt) {
  __shared__ _Float16 bsh[64 * (DIMC + LPAD)];
  const int wave = threadIdx.x >> 5, lane = threadIdx.x & 31;
  const int m0  = (blockIdx.x * 4 + wave) * 16;
  const int n0s = blockIdx.y * 64;
  v8f acc[4];
  gemm_strip<DIMC, true>(A, Wt, m0, n0s, bsh, acc);

  const int col = lane & 15, mb = (lane >> 4) << 3;
  const int j3  = n0s >> 7;                    // uniform: 0=Q,1=K,2=V
  const int b   = m0 / NTOK;                   // uniform: tile never crosses batch
  const int i0  = m0 - b * NTOK + mb;
#pragma unroll
  for (int t = 0; t < 4; ++t) {
    const int hd = ((n0s & 127) >> 4) + t;
    const size_t bh = (size_t)(b * NHEAD + hd);
#pragma unroll
    for (int r = 0; r < 8; ++r) {
      int i = i0 + r;
      _Float16 v = (_Float16)acc[t][r];
      if (j3 == 0)      Qh[(bh * NTOK + i) * DHEAD + col] = v;
      else if (j3 == 1) Kh[(bh * NTOK + i) * DHEAD + col] = v;
      else              Vt[(bh * DHEAD + col) * NTOK + i] = v;
    }
  }
}

// ---------------- flash attention: one wave per (b,h,16-query tile)
__global__ void k_attn(const _Float16* __restrict__ Qh, const _Float16* __restrict__ Kh,
                       const _Float16* __restrict__ Vt, _Float16* __restrict__ attout) {
  const int bh   = blockIdx.x;               // 0..15
  const int wave = threadIdx.x >> 5;
  const int lane = threadIdx.x & 31;
  const int qt   = blockIdx.y * 4 + wave;    // 0..191
  const int b = bh >> 3, hd = bh & 7;
  const int col = lane & 15;
  const int hw  = lane >> 4;
  const int q0  = qt * 16;

  __shared__ _Float16 lds[4 * 256];          // 16x16 f16 P-tile per wave
  _Float16* pt = lds + wave * 256;

  v16h zero = {};
  v16h aq = zero;
  if (hw == 0) aq = *(const v16h*)(Qh + ((size_t)bh * NTOK + q0 + col) * DHEAD);

  float mrun[8], lrun[8];
  v8f o = {};
#pragma unroll
  for (int r = 0; r < 8; ++r) { mrun[r] = -1e30f; lrun[r] = 0.0f; }

  for (int j = 0; j < NKT; ++j) {
    v16h bk = zero;
    if (hw == 0) bk = *(const v16h*)(Kh + ((size_t)bh * NTOK + j * 16 + col) * DHEAD);
    v8f s = {};
    s = __builtin_amdgcn_wmma_f32_16x16x32_f16(false, aq, false, bk, (short)0, s, false, false);
#pragma unroll
    for (int r = 0; r < 8; ++r) {
      float sv   = s[r] * ATT_SCALE;
      float mnew = fmaxf(mrun[r], rmax16(sv));
      float p    = __expf(sv - mnew);
      float corr = __expf(mrun[r] - mnew);
      lrun[r] = lrun[r] * corr + rsum16(p);
      mrun[r] = mnew;
      o[r]    = o[r] * corr;
      pt[((r + (hw << 3)) << 4) + col] = (_Float16)p;   // C-layout -> LDS
    }
    v16h pa = zero, bv = zero;
    if (hw == 0) {
      pa = *(const v16h*)(pt + (col << 4));                                  // A-layout P
      bv = *(const v16h*)(Vt + ((size_t)bh * DHEAD + col) * NTOK + j * 16);  // B-frag of V
    }
    o = __builtin_amdgcn_wmma_f32_16x16x32_f16(false, pa, false, bv, (short)0, o, false, false);
  }
#pragma unroll
  for (int r = 0; r < 8; ++r) {
    int row = q0 + (hw << 3) + r;
    float v = o[r] / lrun[r];
    attout[((size_t)b * NTOK + row) * DIMC + hd * DHEAD + col] = (_Float16)v;
  }
}

// ---------------- GEMM (N=128) + bias + residual accumulate into fp32 h
template<int K>
__global__ void k_gemm_bias_res(const _Float16* __restrict__ A, const _Float16* __restrict__ Wt,
                                const float* __restrict__ bias, float* __restrict__ h) {
  __shared__ _Float16 bsh[(K == DIMC) ? 64 * (DIMC + LPAD) : 16];
  const int wave = threadIdx.x >> 5, lane = threadIdx.x & 31;
  const int m0  = (blockIdx.x * 4 + wave) * 16;
  const int n0s = blockIdx.y * 64;
  v8f acc[4];
  gemm_strip<K, (K == DIMC)>(A, Wt, m0, n0s, bsh, acc);
  const int col = lane & 15, mb = (lane >> 4) << 3;
#pragma unroll
  for (int t = 0; t < 4; ++t) {
    int cg = n0s + t * 16 + col;
    float bc = bias[cg];
#pragma unroll
    for (int r = 0; r < 8; ++r) {
      int mg = m0 + mb + r;
      h[(size_t)mg * DIMC + cg] += acc[t][r] + bc;
    }
  }
}

// ---------------- GEMM + bias + exact GELU -> f16 activation [M,N]
template<int N>
__global__ void k_gemm_bias_gelu(const _Float16* __restrict__ A, const _Float16* __restrict__ Wt,
                                 const float* __restrict__ bias, _Float16* __restrict__ out) {
  __shared__ _Float16 bsh[64 * (DIMC + LPAD)];
  const int wave = threadIdx.x >> 5, lane = threadIdx.x & 31;
  const int m0  = (blockIdx.x * 4 + wave) * 16;
  const int n0s = blockIdx.y * 64;
  v8f acc[4];
  gemm_strip<DIMC, true>(A, Wt, m0, n0s, bsh, acc);
  const int col = lane & 15, mb = (lane >> 4) << 3;
#pragma unroll
  for (int t = 0; t < 4; ++t) {
    int cg = n0s + t * 16 + col;
    float bc = bias[cg];
#pragma unroll
    for (int r = 0; r < 8; ++r) {
      int mg = m0 + mb + r;
      float xv = acc[t][r] + bc;
      float ge = 0.5f * xv * (1.0f + erff(xv * 0.70710678118f));
      out[(size_t)mg * N + cg] = (_Float16)ge;
    }
  }
}

// ---------------- head: one wave per output scalar
__global__ void k_head(const float* __restrict__ h, const float* __restrict__ wout,
                       const float* __restrict__ bout, float* __restrict__ out) {
  int wave = threadIdx.x >> 5, lane = threadIdx.x & 31;
  int o = blockIdx.x * 8 + wave;        // 0..511
  int b = o >> 8, p = o & 255;
  int row = b * NTOK + 11 * 256 + p;
  float4 v = ((const float4*)(h + (size_t)row * DIMC))[lane];
  float4 w = ((const float4*)wout)[lane];
  float s = v.x * w.x + v.y * w.y + v.z * w.z + v.w * w.w;
  s = rsum32(s);
  if (lane == 0) out[o] = s + bout[0];
}

// ================= host =================
static inline size_t alup(size_t x) { return (x + 255) & ~(size_t)255; }

extern "C" void kernel_launch(void* const* d_in, const int* in_sizes, int n_in,
                              void* d_out, int out_size, void* d_ws, size_t ws_size,
                              hipStream_t stream) {
  (void)in_sizes; (void)n_in; (void)out_size; (void)ws_size;
  const float* x       = (const float*)d_in[0];
  const float* pos_emb = (const float*)d_in[1];
  const float* time_emb= (const float*)d_in[2];
  const float* w_patch = (const float*)d_in[3];
  const float* b_patch = (const float*)d_in[4];
  const float* ln1_g   = (const float*)d_in[5];
  const float* ln1_b   = (const float*)d_in[6];
  const float* w_qkv   = (const float*)d_in[7];
  const float* w_o     = (const float*)d_in[8];
  const float* b_o     = (const float*)d_in[9];
  const float* ln2_g   = (const float*)d_in[10];
  const float* ln2_b   = (const float*)d_in[11];
  const float* w1      = (const float*)d_in[12];
  const float* b1      = (const float*)d_in[13];
  const float* w2      = (const float*)d_in[14];
  const float* b2      = (const float*)d_in[15];
  const float* w_out   = (const float*)d_in[16];
  const float* b_out   = (const float*)d_in[17];

  char* ws = (char*)d_ws;
  size_t off = 0;
  float*    hbuf  = (float*)(ws + off);    off += alup((size_t)ROWS * DIMC * 4);
  _Float16* a16   = (_Float16*)(ws + off); off += alup((size_t)ROWS * DIMC * 2);
  _Float16* Qh    = (_Float16*)(ws + off); off += alup((size_t)2 * NHEAD * NTOK * DHEAD * 2);
  _Float16* Kh    = (_Float16*)(ws + off); off += alup((size_t)2 * NHEAD * NTOK * DHEAD * 2);
  _Float16* Vt    = (_Float16*)(ws + off); off += alup((size_t)2 * NHEAD * NTOK * DHEAD * 2);
  _Float16* att16 = (_Float16*)(ws + off); off += alup((size_t)ROWS * DIMC * 2);
  _Float16* act16 = (_Float16*)(ws + off); off += alup((size_t)ROWS * MLPD * 2);
  _Float16* wqkvT = (_Float16*)(ws + off); off += alup((size_t)NDEPTH * 384 * DIMC * 2);
  _Float16* woT   = (_Float16*)(ws + off); off += alup((size_t)NDEPTH * DIMC * DIMC * 2);
  _Float16* w1T   = (_Float16*)(ws + off); off += alup((size_t)NDEPTH * MLPD * DIMC * 2);
  _Float16* w2T   = (_Float16*)(ws + off); off += alup((size_t)NDEPTH * DIMC * MLPD * 2);

  {
    int t;
    t = NDEPTH * DIMC * 384;  k_wconv_t<<<(t + 255) / 256, 256, 0, stream>>>(w_qkv, wqkvT, DIMC, 384, t);
    t = NDEPTH * DIMC * DIMC; k_wconv_t<<<(t + 255) / 256, 256, 0, stream>>>(w_o, woT, DIMC, DIMC, t);
    t = NDEPTH * DIMC * MLPD; k_wconv_t<<<(t + 255) / 256, 256, 0, stream>>>(w1, w1T, DIMC, MLPD, t);
    t = NDEPTH * MLPD * DIMC; k_wconv_t<<<(t + 255) / 256, 256, 0, stream>>>(w2, w2T, MLPD, DIMC, t);
  }

  k_patch_embed<<<ROWS, DIMC, 0, stream>>>(x, pos_emb, time_emb, w_patch, b_patch, hbuf);

  for (int l = 0; l < NDEPTH; ++l) {
    k_ln<<<ROWS / 8, 256, 0, stream>>>(hbuf, ln1_g + l * DIMC, ln1_b + l * DIMC, a16);
    k_gemm_qkv<<<dim3(ROWS / 64, 6), 128, 0, stream>>>(a16, wqkvT + (size_t)l * 384 * DIMC, Qh, Kh, Vt);
    k_attn<<<dim3(16, 48), 128, 0, stream>>>(Qh, Kh, Vt, att16);
    k_gemm_bias_res<DIMC><<<dim3(ROWS / 64, 2), 128, 0, stream>>>(att16, woT + (size_t)l * DIMC * DIMC, b_o + l * DIMC, hbuf);
    k_ln<<<ROWS / 8, 256, 0, stream>>>(hbuf, ln2_g + l * DIMC, ln2_b + l * DIMC, a16);
    k_gemm_bias_gelu<MLPD><<<dim3(ROWS / 64, 8), 128, 0, stream>>>(a16, w1T + (size_t)l * MLPD * DIMC, b1 + l * MLPD, act16);
    k_gemm_bias_res<MLPD><<<dim3(ROWS / 64, 2), 128, 0, stream>>>(act16, w2T + (size_t)l * DIMC * MLPD, b2 + l * DIMC, hbuf);
  }

  k_head<<<64, 256, 0, stream>>>(hbuf, w_out, b_out, (float*)d_out);
}
